// CardHandEmbed_19971597927146
// MI455X (gfx1250) — compile-verified, matrix operand order checked by reading, not run on hardware
//
#include <hip/hip_runtime.h>
#include <hip/hip_bf16.h>

#define BQ 64
#define NQ 128
#define M_CARDS (BQ*NQ)          /* 8192 */
#define ID_DIM 256
#define TYPE_DIM 64
#define MANA_DIM 32
#define N_COST 6
#define SP_LEN 32
#define CAT_DIM 128
#define CONT_HID 64
#define EMBED 512
#define HEAD_IN 640
#define N_HEADS 9
#define FUSED_IN 768
#define EPSF 1e-6f
#define MTILE 64                 /* cards per workgroup */

typedef __attribute__((ext_vector_type(16))) _Float16 v16h;
typedef __attribute__((ext_vector_type(8)))  _Float16 v8h;
typedef __attribute__((ext_vector_type(8)))  float    v8f;
typedef __attribute__((ext_vector_type(4)))  unsigned int u32x4;
typedef __attribute__((ext_vector_type(8)))  int i32x8;
typedef __attribute__((ext_vector_type(4)))  int i32x4;

// ---------------------------------------------------------------------------
// K1: convert Wh [9][640][512] -> whT f16 [9][512][640], Wf [768][512] -> wfT f16 [512][768]
// ---------------------------------------------------------------------------
__global__ void k_convert(const float* __restrict__ Wh, const float* __restrict__ Wf,
                          _Float16* __restrict__ whT, _Float16* __restrict__ wfT) {
  const int nWh = N_HEADS * HEAD_IN * EMBED;
  const int nWf = FUSED_IN * EMBED;
  for (int idx = blockIdx.x * blockDim.x + threadIdx.x; idx < nWh + nWf;
       idx += gridDim.x * blockDim.x) {
    if (idx < nWh) {
      int t   = idx / (HEAD_IN * EMBED);
      int rem = idx - t * (HEAD_IN * EMBED);
      int o   = rem / HEAD_IN;
      int i   = rem - o * HEAD_IN;
      whT[idx] = (_Float16)Wh[(size_t)t * HEAD_IN * EMBED + (size_t)i * EMBED + o];
    } else {
      int j = idx - nWh;
      int o = j / FUSED_IN;
      int k = j - o * FUSED_IN;
      wfT[j] = (_Float16)Wf[(size_t)k * EMBED + o];
    }
  }
}

// ---------------------------------------------------------------------------
// K2: featurize — one wave32 per card. Builds head_in f16 [8192][640] and
// cont f16 [8192][256] (gathers + tiny MLPs done with VALU + wave shuffles).
// ---------------------------------------------------------------------------
__global__ __launch_bounds__(128) void k_featurize(
    const int* __restrict__ card_id, const int* __restrict__ card_type,
    const int* __restrict__ card_cost, const float* __restrict__ sp,
    const float* __restrict__ atk_n, const float* __restrict__ hp_n,
    const float* __restrict__ has_atk, const float* __restrict__ has_hp,
    const float* __restrict__ id_table, const float* __restrict__ type_table,
    const float* __restrict__ cost_table,
    const float* __restrict__ Wsp, const float* __restrict__ bsp,
    const float* __restrict__ Wc1, const float* __restrict__ bc1,
    const float* __restrict__ Wc2, const float* __restrict__ bc2,
    _Float16* __restrict__ hin, _Float16* __restrict__ cont) {
  const int lane = threadIdx.x & 31;
  const int card = blockIdx.x * 4 + (threadIdx.x >> 5);
  const int cid = card_id[card];
  const int ct  = card_type[card];
  _Float16* hrow = hin + (size_t)card * HEAD_IN;

  for (int c = lane; c < ID_DIM; c += 32)
    hrow[c] = (_Float16)id_table[(size_t)cid * ID_DIM + c];
  for (int c = lane; c < TYPE_DIM; c += 32)
    hrow[ID_DIM + c] = (_Float16)type_table[ct * TYPE_DIM + c];
  float mana = 0.f;
#pragma unroll
  for (int j = 0; j < N_COST; ++j) {
    int cc = card_cost[card * N_COST + j];
    mana += (float)cc;
    hrow[ID_DIM + TYPE_DIM + j * MANA_DIM + lane] =
        (_Float16)cost_table[cc * MANA_DIM + lane];
  }
  float spv = sp[card * SP_LEN + lane];
  for (int c = lane; c < CAT_DIM; c += 32) {
    float acc = bsp[c];
#pragma unroll
    for (int k = 0; k < SP_LEN; ++k)
      acc += __shfl(spv, k, 32) * Wsp[k * CAT_DIM + c];
    hrow[ID_DIM + TYPE_DIM + MANA_DIM * N_COST + c] = (_Float16)tanhf(acc);
  }
  float atk = atk_n[card], hp = hp_n[card], ha = has_atk[card], hh = has_hp[card];
  float inv = 1.f / (mana + EPSF);
  float cin[9] = {mana, atk, hp, ha, hh, atk + hp, atk * hp, atk * inv, (atk + hp) * inv};
  float h1a = bc1[lane], h1b = bc1[lane + 32];
#pragma unroll
  for (int k = 0; k < 9; ++k) {
    h1a += cin[k] * Wc1[k * CONT_HID + lane];
    h1b += cin[k] * Wc1[k * CONT_HID + lane + 32];
  }
  h1a = tanhf(h1a);
  h1b = tanhf(h1b);
  for (int c = lane; c < ID_DIM; c += 32) {
    float acc = bc2[c];
#pragma unroll 8
    for (int k = 0; k < 32; ++k) acc += __shfl(h1a, k, 32) * Wc2[k * ID_DIM + c];
#pragma unroll 8
    for (int k = 0; k < 32; ++k) acc += __shfl(h1b, k, 32) * Wc2[(k + 32) * ID_DIM + c];
    cont[(size_t)card * ID_DIM + c] = (_Float16)tanhf(acc);
  }
}

// ---------------------------------------------------------------------------
// K3: head GEMM. One workgroup = 64 cards x 64 cols (4 waves x 16 cols).
// A tile (64x640 f16 = 80KB dynamic LDS) staged via TDM tensor_load_to_lds;
// each B fragment feeds 4 WMMAs (4 M sub-tiles) -> 4x less L2 weight traffic.
// 9 heads x 20 K-steps x 4 subtiles = 720 WMMAs/wave.
// ---------------------------------------------------------------------------
__global__ __launch_bounds__(128) void k_heads(
    const _Float16* __restrict__ hin, const _Float16* __restrict__ whT,
    const float* __restrict__ bh, const int* __restrict__ card_type,
    _Float16* __restrict__ typed) {
  extern __shared__ __align__(16) char smem[];
  _Float16* As  = (_Float16*)smem;                     // 64 x 640 f16 = 81920 B
  int* tlds     = (int*)(smem + MTILE * HEAD_IN * 2);  // 64 ints
  const int tid = threadIdx.x;
  const int cardb = blockIdx.x * MTILE;

  if (tid == 0) {
    // --- Tensor DMA descriptor (D#): 2D tile, data_size=4B ---
    // rows: 64, row length: 640 f16 = 320 dwords, contiguous (stride == dim0)
    unsigned long long ga = (unsigned long long)(const void*)(hin + (size_t)cardb * HEAD_IN);
    unsigned ldsoff = (unsigned)(unsigned long long)(&As[0]);
    u32x4 g0 = { 1u,                                   // count=1, user mode
                 ldsoff,                               // lds_addr
                 (unsigned)ga,                         // global_addr[31:0]
                 ((unsigned)(ga >> 32) & 0x01FFFFFFu) | (2u << 30) };  // addr[56:32] | type=2
    i32x8 g1 = { (int)(2u << 16),        // data_size = 2 (4 bytes)
                 (int)(320u << 16),      // tensor_dim0 = 320 dwords
                 (int)(64u  << 16),      // tensor_dim1 = 64
                 (int)(320u << 16),      // tile_dim0 = 320
                 (int)64,                // tile_dim1 = 64, tile_dim2 = 0
                 (int)320,               // tensor_dim0_stride = 320
                 0, 0 };
    i32x4 gz4 = { 0, 0, 0, 0 };
    i32x8 gz8 = { 0, 0, 0, 0, 0, 0, 0, 0 };
    __builtin_amdgcn_tensor_load_to_lds(g0, g1, gz4, gz4, gz8, 0);
  }
  if (tid < MTILE) tlds[tid] = card_type[cardb + tid];
  if (tid == 0) __builtin_amdgcn_s_wait_tensorcnt(0);
  __syncthreads();

  const int lane  = tid & 31;
  const int wave  = tid >> 5;
  const int nlane = lane & 15;
  const int kh    = lane >> 4;
  const int kh8   = kh * 8;
  const int col   = (blockIdx.y * 4 + wave) * 16 + nlane;

  const _Float16* arow[4];
#pragma unroll
  for (int m = 0; m < 4; ++m) arow[m] = &As[(unsigned)(m * 16 + nlane) * HEAD_IN];

  float sel[4][8];
#pragma unroll
  for (int m = 0; m < 4; ++m)
#pragma unroll
    for (int i = 0; i < 8; ++i) sel[m][i] = 0.f;

  for (int t = 0; t < N_HEADS; ++t) {
    v8f acc[4];
#pragma unroll
    for (int m = 0; m < 4; ++m) acc[m] = (v8f){};
    const _Float16* bp = whT + ((size_t)t * EMBED + col) * HEAD_IN;
#pragma unroll 2
    for (int kk = 0; kk < HEAD_IN / 32; ++kk) {
      const int kb = kk * 32;
      v8h blo = *(const v8h*)(bp + kb + kh8);
      v8h bhi = *(const v8h*)(bp + kb + 16 + kh8);
      __builtin_prefetch(bp + kb + 256, 0, 3);
      v16h b;
#pragma unroll
      for (int e = 0; e < 8; ++e) { b[e] = blo[e]; b[e + 8] = bhi[e]; }
#pragma unroll
      for (int m = 0; m < 4; ++m) {
        v8h alo = *(const v8h*)(arow[m] + kb + kh8);
        v8h ahi = *(const v8h*)(arow[m] + kb + 16 + kh8);
        v16h a;
#pragma unroll
        for (int e = 0; e < 8; ++e) { a[e] = alo[e]; a[e + 8] = ahi[e]; }
        acc[m] = __builtin_amdgcn_wmma_f32_16x16x32_f16(
            false, a, false, b, (short)0, acc[m], false, false);
      }
    }
    const float bias = bh[t * EMBED + col];
#pragma unroll
    for (int m = 0; m < 4; ++m)
#pragma unroll
      for (int i = 0; i < 8; ++i) {
        int r = m * 16 + i + kh8;
        sel[m][i] = (tlds[r] == t) ? (acc[m][i] + bias) : sel[m][i];
      }
  }
#pragma unroll
  for (int m = 0; m < 4; ++m)
#pragma unroll
    for (int i = 0; i < 8; ++i) {
      int r = m * 16 + i + kh8;
      typed[(size_t)(cardb + r) * EMBED + col] = (_Float16)tanhf(sel[m][i]);
    }
}

// ---------------------------------------------------------------------------
// K4: fused GEMM. 64 cards x 64 cols per workgroup. A split into a typed
// block [64][512] and cont block [64][256] in dynamic LDS (96KB), staged via
// gfx1250 async-to-LDS (pure linear copies). K-loop split 16 + 8 steps.
// ---------------------------------------------------------------------------
__global__ __launch_bounds__(128) void k_fused(
    const _Float16* __restrict__ typed, const _Float16* __restrict__ cont,
    const _Float16* __restrict__ wfT, const float* __restrict__ bf,
    float* __restrict__ out) {
  extern __shared__ __align__(16) char smem[];
  _Float16* At = (_Float16*)smem;                       // 64 x 512 f16 = 65536 B
  _Float16* Ac = (_Float16*)(smem + MTILE * EMBED * 2); // 64 x 256 f16 = 32768 B
  const int tid = threadIdx.x;
  const int cardb = blockIdx.x * MTILE;
  {
    const _Float16* g0 = typed + (size_t)cardb * EMBED;
    unsigned lb0 = (unsigned)(unsigned long long)(&At[0]);
#pragma unroll
    for (int it = 0; it < 32; ++it) {                   // 65536 B linear
      int off = (it * 128 + tid) * 16;
      unsigned laddr = lb0 + (unsigned)off;
      asm volatile("global_load_async_to_lds_b128 %0, %1, %2"
                   :: "v"(laddr), "v"(off), "s"(g0) : "memory");
    }
    const _Float16* g1 = cont + (size_t)cardb * ID_DIM;
    unsigned lb1 = (unsigned)(unsigned long long)(&Ac[0]);
#pragma unroll
    for (int it = 0; it < 16; ++it) {                   // 32768 B linear
      int off = (it * 128 + tid) * 16;
      unsigned laddr = lb1 + (unsigned)off;
      asm volatile("global_load_async_to_lds_b128 %0, %1, %2"
                   :: "v"(laddr), "v"(off), "s"(g1) : "memory");
    }
  }
  asm volatile("s_wait_asynccnt 0" ::: "memory");
  __syncthreads();

  const int lane  = tid & 31;
  const int wave  = tid >> 5;
  const int nlane = lane & 15;
  const int kh    = lane >> 4;
  const int kh8   = kh * 8;
  const int col   = (blockIdx.y * 4 + wave) * 16 + nlane;

  const _Float16* bp = wfT + (size_t)col * FUSED_IN;
  v8f acc[4];
#pragma unroll
  for (int m = 0; m < 4; ++m) acc[m] = (v8f){};

#pragma unroll 2
  for (int kk = 0; kk < EMBED / 32; ++kk) {             // typed block: 16 steps
    const int kb = kk * 32;
    v8h blo = *(const v8h*)(bp + kb + kh8);
    v8h bhi = *(const v8h*)(bp + kb + 16 + kh8);
    v16h b;
#pragma unroll
    for (int e = 0; e < 8; ++e) { b[e] = blo[e]; b[e + 8] = bhi[e]; }
#pragma unroll
    for (int m = 0; m < 4; ++m) {
      const _Float16* ar = At + (unsigned)(m * 16 + nlane) * EMBED;
      v8h alo = *(const v8h*)(ar + kb + kh8);
      v8h ahi = *(const v8h*)(ar + kb + 16 + kh8);
      v16h a;
#pragma unroll
      for (int e = 0; e < 8; ++e) { a[e] = alo[e]; a[e + 8] = ahi[e]; }
      acc[m] = __builtin_amdgcn_wmma_f32_16x16x32_f16(
          false, a, false, b, (short)0, acc[m], false, false);
    }
  }
#pragma unroll 2
  for (int kk = 0; kk < ID_DIM / 32; ++kk) {            // cont block: 8 steps
    const int kb = kk * 32;
    v8h blo = *(const v8h*)(bp + EMBED + kb + kh8);
    v8h bhi = *(const v8h*)(bp + EMBED + kb + 16 + kh8);
    v16h b;
#pragma unroll
    for (int e = 0; e < 8; ++e) { b[e] = blo[e]; b[e + 8] = bhi[e]; }
#pragma unroll
    for (int m = 0; m < 4; ++m) {
      const _Float16* ar = Ac + (unsigned)(m * 16 + nlane) * ID_DIM;
      v8h alo = *(const v8h*)(ar + kb + kh8);
      v8h ahi = *(const v8h*)(ar + kb + 16 + kh8);
      v16h a;
#pragma unroll
      for (int e = 0; e < 8; ++e) { a[e] = alo[e]; a[e + 8] = ahi[e]; }
      acc[m] = __builtin_amdgcn_wmma_f32_16x16x32_f16(
          false, a, false, b, (short)0, acc[m], false, false);
    }
  }
  const float bias = bf[col];
#pragma unroll
  for (int m = 0; m < 4; ++m)
#pragma unroll
    for (int i = 0; i < 8; ++i) {
      int r = m * 16 + i + kh8;
      out[(size_t)(cardb + r) * EMBED + col] = tanhf(acc[m][i] + bias);
    }
}

// ---------------------------------------------------------------------------
extern "C" void kernel_launch(void* const* d_in, const int* in_sizes, int n_in,
                              void* d_out, int out_size, void* d_ws, size_t ws_size,
                              hipStream_t stream) {
  const int*   card_id    = (const int*)d_in[0];
  const int*   card_type  = (const int*)d_in[1];
  const int*   card_cost  = (const int*)d_in[2];
  const float* sp         = (const float*)d_in[3];
  const float* atk_n      = (const float*)d_in[4];
  const float* hp_n       = (const float*)d_in[5];
  const float* has_atk    = (const float*)d_in[6];
  const float* has_hp     = (const float*)d_in[7];
  const float* id_table   = (const float*)d_in[8];
  const float* type_table = (const float*)d_in[9];
  const float* cost_table = (const float*)d_in[10];
  const float* Wsp = (const float*)d_in[11];
  const float* bsp = (const float*)d_in[12];
  const float* Wc1 = (const float*)d_in[13];
  const float* bc1 = (const float*)d_in[14];
  const float* Wc2 = (const float*)d_in[15];
  const float* bc2 = (const float*)d_in[16];
  const float* Wh  = (const float*)d_in[17];
  const float* bh  = (const float*)d_in[18];
  const float* Wf  = (const float*)d_in[19];
  const float* bf  = (const float*)d_in[20];

  char* ws = (char*)d_ws;
  _Float16* hin16   = (_Float16*)(ws);                 // 8192*640*2  = 10485760
  _Float16* cont16  = (_Float16*)(ws + 10485760);      // 8192*256*2  =  4194304
  _Float16* typed16 = (_Float16*)(ws + 14680064);      // 8192*512*2  =  8388608
  _Float16* whT     = (_Float16*)(ws + 23068672);      // 9*640*512*2 =  5898240
  _Float16* wfT     = (_Float16*)(ws + 28966912);      // 768*512*2   =   786432

  k_convert<<<1024, 256, 0, stream>>>(Wh, Wf, whT, wfT);
  k_featurize<<<M_CARDS / 4, 128, 0, stream>>>(
      card_id, card_type, card_cost, sp, atk_n, hp_n, has_atk, has_hp,
      id_table, type_table, cost_table, Wsp, bsp, Wc1, bc1, Wc2, bc2,
      hin16, cont16);
  k_heads<<<dim3(M_CARDS / MTILE, 8), 128, MTILE * HEAD_IN * 2 + 256, stream>>>(
      hin16, whT, bh, card_type, typed16);
  k_fused<<<dim3(M_CARDS / MTILE, 8), 128, MTILE * FUSED_IN * 2, stream>>>(
      typed16, cont16, wfT, bf, (float*)d_out);
}
